// EncoderDecoderLSTMCell_45887430590720
// MI455X (gfx1250) — compile-verified, matrix operand order checked by reading, not run on hardware
//
#include <hip/hip_runtime.h>

// ---------------------------------------------------------------------------
// EncoderDecoderLSTM on MI455X (gfx1250, wave32, WMMA bf16).
// Strategy:
//   prep    : convert weights fp32->bf16 (weights stay hot in 192MB L2),
//             fuse b_ih+b_hh, zero h0/c0.
//   enc     : encoded = input @ W_enc^T + b_enc   (WMMA, fully parallel)
//   step×640: gates = x@W_ih^T + h@W_hh^T + bias; c,h update (WMMA + VALU).
//             One kernel launch per timestep -> stream order carries the
//             recurrence dependency. h_t written into hs[t] (bf16), read as
//             h_{t-1} by the next launch (natural double buffering).
//   fenc×128: u = h @ W_fenc^T + b_fenc for future steps (WMMA).
//   dec     : out = hs @ W_dec^T + b_dec over all 81920 rows (WMMA).
// ---------------------------------------------------------------------------

typedef __bf16 bf16_t;
typedef __attribute__((ext_vector_type(16))) __bf16 v16bf;
typedef __attribute__((ext_vector_type(8)))  __bf16 bf16x8;
typedef __attribute__((ext_vector_type(8)))  float  v8f;

#define B_    128
#define T_    512
#define FUT_  128
#define FIN_  64
#define RIP_  256
#define HID_  512
#define G_    2048   // 4*HID
#define FOUT_ 64

__device__ __forceinline__ v8f wmma_bf16(v16bf a, v16bf b, v8f c) {
  return __builtin_amdgcn_wmma_f32_16x16x32_bf16(false, a, false, b,
                                                 (short)0, c, false, false);
}

// 16x32 bf16 fragment from a row-major tile (row stride in elements).
// ISA layout: lanes 0-15 -> row=lane, K {0..7,16..23}; lanes 16-31 -> K {8..15,24..31}.
__device__ __forceinline__ v16bf load_frag_bf16(const bf16_t* __restrict__ tile,
                                                int stride, int lane) {
  int r = lane & 15;
  const bf16_t* p = tile + r * stride + ((lane >> 4) << 3);
  bf16x8 lo = *reinterpret_cast<const bf16x8*>(p);
  bf16x8 hi = *reinterpret_cast<const bf16x8*>(p + 16);
  return __builtin_shufflevector(lo, hi, 0,1,2,3,4,5,6,7,8,9,10,11,12,13,14,15);
}

// Same fragment, sourced from fp32 memory (converted on the fly).
__device__ __forceinline__ v16bf load_frag_f32(const float* __restrict__ tile,
                                               int stride, int lane) {
  int r = lane & 15;
  const float* p = tile + r * stride + ((lane >> 4) << 3);
  v16bf f;
#pragma unroll
  for (int i = 0; i < 8; ++i) {
    f[i]     = (bf16_t)p[i];
    f[i + 8] = (bf16_t)p[i + 16];
  }
  return f;
}

__device__ __forceinline__ float sigf(float x) {
  return 1.0f / (1.0f + __expf(-x));   // exp(inf)->inf -> 0, no NaN
}
__device__ __forceinline__ float tanhfast(float x) {
  float e = __expf(2.0f * fabsf(x));   // overflow -> inf -> r = 1
  float r = 1.0f - 2.0f / (e + 1.0f);
  return copysignf(r, x);
}

// ---------------------------------------------------------------------------
// prep: weight conversion, bias fusion, state zeroing
// ---------------------------------------------------------------------------
__global__ __launch_bounds__(256) void prep_kernel(
    const float* __restrict__ W_enc, const float* __restrict__ W_ih,
    const float* __restrict__ W_hh,  const float* __restrict__ W_fenc,
    const float* __restrict__ W_dec, const float* __restrict__ b_ih,
    const float* __restrict__ b_hh,
    bf16_t* __restrict__ Wenc_bf, bf16_t* __restrict__ Wih_bf,
    bf16_t* __restrict__ Whh_bf,  bf16_t* __restrict__ Wfenc_bf,
    bf16_t* __restrict__ Wdec_bf, float* __restrict__ bsum,
    float* __restrict__ cbuf, bf16_t* __restrict__ h0) {
  int i = blockIdx.x * 256 + threadIdx.x;
  int st = gridDim.x * 256;
  for (int k = i; k < G_ * HID_;   k += st) Whh_bf[k]   = (bf16_t)W_hh[k];
  for (int k = i; k < G_ * RIP_;   k += st) Wih_bf[k]   = (bf16_t)W_ih[k];
  for (int k = i; k < RIP_ * FIN_; k += st) Wenc_bf[k]  = (bf16_t)W_enc[k];
  for (int k = i; k < RIP_ * HID_; k += st) Wfenc_bf[k] = (bf16_t)W_fenc[k];
  for (int k = i; k < FOUT_ * HID_;k += st) Wdec_bf[k]  = (bf16_t)W_dec[k];
  for (int k = i; k < G_;          k += st) bsum[k]     = b_ih[k] + b_hh[k];
  for (int k = i; k < B_ * HID_;   k += st) { cbuf[k] = 0.0f; h0[k] = (bf16_t)0.0f; }
}

// ---------------------------------------------------------------------------
// encoder: encoded[b*T+t][0..255] = input[b*T+t][0..63] @ W_enc^T + b_enc
// one 16x16 tile per wave; M=65536 rows, N=256, K=64
// ---------------------------------------------------------------------------
__global__ __launch_bounds__(256) void enc_kernel(
    const float* __restrict__ X, const bf16_t* __restrict__ W,
    const float* __restrict__ bias, bf16_t* __restrict__ Y) {
  int wave = (blockIdx.x * 256 + threadIdx.x) >> 5;
  int lane = threadIdx.x & 31;
  int mt = wave >> 4;     // 0..4095
  int nt = wave & 15;     // 0..15
  v8f acc = {0,0,0,0,0,0,0,0};
  const float*  Ab = X + (size_t)mt * 16 * FIN_;
  const bf16_t* Bb = W + nt * 16 * FIN_;
#pragma unroll
  for (int kp = 0; kp < FIN_ / 32; ++kp)
    acc = wmma_bf16(load_frag_f32(Ab + kp * 32, FIN_, lane),
                    load_frag_bf16(Bb + kp * 32, FIN_, lane), acc);
  int col = nt * 16 + (lane & 15);
  float bv = bias[col];
  int rbase = mt * 16 + ((lane >> 4) << 3);   // C layout: lanes16-31 -> M+8
#pragma unroll
  for (int r = 0; r < 8; ++r)
    Y[(size_t)(rbase + r) * RIP_ + col] = (bf16_t)(acc[r] + bv);
}

// ---------------------------------------------------------------------------
// one LSTM timestep. Each wave owns (m: 16 batch rows, jt: 16 hidden cols)
// and computes the i/f/g/o gate tiles for that slice, then the c/h update.
// x: [128][256] (row stride parameterized: strided view into encoded, or u).
// h_in: previous hs slice (bf16, stride 512). h_out -> hs[t].
// ---------------------------------------------------------------------------
__global__ __launch_bounds__(256) void lstm_step_kernel(
    const bf16_t* __restrict__ x_base, int x_stride,
    const bf16_t* __restrict__ Wih, const bf16_t* __restrict__ Whh,
    const float* __restrict__ bsum, float* __restrict__ cbuf,
    const bf16_t* __restrict__ h_in, bf16_t* __restrict__ h_out) {
  int wave = (blockIdx.x * 256 + threadIdx.x) >> 5;   // 0..255
  int lane = threadIdx.x & 31;
  int m  = wave >> 5;   // 0..7  (batch tile)
  int jt = wave & 31;   // 0..31 (hidden tile)
  v8f ai = {0,0,0,0,0,0,0,0}, af = ai, ag = ai, ao = ai;

  // x @ W_ih^T  (K = 256)
  {
    const bf16_t* Ab = x_base + (size_t)m * 16 * x_stride;
    const bf16_t* B0 = Wih + (size_t)(0 * HID_ + jt * 16) * RIP_;
    const bf16_t* B1 = Wih + (size_t)(1 * HID_ + jt * 16) * RIP_;
    const bf16_t* B2 = Wih + (size_t)(2 * HID_ + jt * 16) * RIP_;
    const bf16_t* B3 = Wih + (size_t)(3 * HID_ + jt * 16) * RIP_;
#pragma unroll
    for (int kp = 0; kp < RIP_ / 32; ++kp) {
      v16bf a = load_frag_bf16(Ab + kp * 32, x_stride, lane);
      ai = wmma_bf16(a, load_frag_bf16(B0 + kp * 32, RIP_, lane), ai);
      af = wmma_bf16(a, load_frag_bf16(B1 + kp * 32, RIP_, lane), af);
      ag = wmma_bf16(a, load_frag_bf16(B2 + kp * 32, RIP_, lane), ag);
      ao = wmma_bf16(a, load_frag_bf16(B3 + kp * 32, RIP_, lane), ao);
    }
  }
  // h @ W_hh^T  (K = 512)
  {
    const bf16_t* Ab = h_in + (size_t)m * 16 * HID_;
    const bf16_t* B0 = Whh + (size_t)(0 * HID_ + jt * 16) * HID_;
    const bf16_t* B1 = Whh + (size_t)(1 * HID_ + jt * 16) * HID_;
    const bf16_t* B2 = Whh + (size_t)(2 * HID_ + jt * 16) * HID_;
    const bf16_t* B3 = Whh + (size_t)(3 * HID_ + jt * 16) * HID_;
#pragma unroll
    for (int kp = 0; kp < HID_ / 32; ++kp) {
      v16bf a = load_frag_bf16(Ab + kp * 32, HID_, lane);
      ai = wmma_bf16(a, load_frag_bf16(B0 + kp * 32, HID_, lane), ai);
      af = wmma_bf16(a, load_frag_bf16(B1 + kp * 32, HID_, lane), af);
      ag = wmma_bf16(a, load_frag_bf16(B2 + kp * 32, HID_, lane), ag);
      ao = wmma_bf16(a, load_frag_bf16(B3 + kp * 32, HID_, lane), ao);
    }
  }

  // elementwise c/h update for this wave's exclusive 16x16 slice
  int col = jt * 16 + (lane & 15);
  float bi = bsum[0 * HID_ + col];
  float bf = bsum[1 * HID_ + col];
  float bg = bsum[2 * HID_ + col];
  float bo = bsum[3 * HID_ + col];
  int rbase = m * 16 + ((lane >> 4) << 3);
#pragma unroll
  for (int r = 0; r < 8; ++r) {
    int row = rbase + r;
    size_t idx = (size_t)row * HID_ + col;
    float iv = sigf(ai[r] + bi);
    float fv = sigf(af[r] + bf);
    float gv = tanhfast(ag[r] + bg);
    float ov = sigf(ao[r] + bo);
    float cv = fv * cbuf[idx] + iv * gv;
    cbuf[idx] = cv;
    h_out[idx] = (bf16_t)(ov * tanhfast(cv));
  }
}

// ---------------------------------------------------------------------------
// future encoder: u = h @ W_fenc^T + b_fenc   (M=128, N=256, K=512)
// ---------------------------------------------------------------------------
__global__ __launch_bounds__(256) void fenc_kernel(
    const bf16_t* __restrict__ h_in, const bf16_t* __restrict__ W,
    const float* __restrict__ bias, bf16_t* __restrict__ u) {
  int wave = (blockIdx.x * 256 + threadIdx.x) >> 5;   // 0..127
  int lane = threadIdx.x & 31;
  int mt = wave >> 4;   // 0..7
  int nt = wave & 15;   // 0..15
  v8f acc = {0,0,0,0,0,0,0,0};
  const bf16_t* Ab = h_in + (size_t)mt * 16 * HID_;
  const bf16_t* Bb = W + (size_t)nt * 16 * HID_;
#pragma unroll
  for (int kp = 0; kp < HID_ / 32; ++kp)
    acc = wmma_bf16(load_frag_bf16(Ab + kp * 32, HID_, lane),
                    load_frag_bf16(Bb + kp * 32, HID_, lane), acc);
  int col = nt * 16 + (lane & 15);
  float bv = bias[col];
  int rbase = mt * 16 + ((lane >> 4) << 3);
#pragma unroll
  for (int r = 0; r < 8; ++r)
    u[(size_t)(rbase + r) * RIP_ + col] = (bf16_t)(acc[r] + bv);
}

// ---------------------------------------------------------------------------
// decoder: out[b][t][:] = hs[t*128+b][:] @ W_dec^T + b_dec
// M = 640*128 = 81920, N = 64, K = 512
// ---------------------------------------------------------------------------
__global__ __launch_bounds__(256) void dec_kernel(
    const bf16_t* __restrict__ hs, const bf16_t* __restrict__ W,
    const float* __restrict__ bias, float* __restrict__ out) {
  int wave = (blockIdx.x * 256 + threadIdx.x) >> 5;
  int lane = threadIdx.x & 31;
  int mt = wave >> 2;   // 0..5119
  int nt = wave & 3;    // 0..3
  v8f acc = {0,0,0,0,0,0,0,0};
  const bf16_t* Ab = hs + (size_t)mt * 16 * HID_;
  const bf16_t* Bb = W + (size_t)nt * 16 * HID_;
#pragma unroll
  for (int kp = 0; kp < HID_ / 32; ++kp)
    acc = wmma_bf16(load_frag_bf16(Ab + kp * 32, HID_, lane),
                    load_frag_bf16(Bb + kp * 32, HID_, lane), acc);
  int col = nt * 16 + (lane & 15);
  float bv = bias[col];
  int Rbase = mt * 16 + ((lane >> 4) << 3);
#pragma unroll
  for (int r = 0; r < 8; ++r) {
    int R = Rbase + r;           // R = t*128 + b
    int t = R >> 7;
    int b = R & 127;
    out[((size_t)b * (T_ + FUT_) + t) * FOUT_ + col] = acc[r] + bv;
  }
}

// ---------------------------------------------------------------------------
extern "C" void kernel_launch(void* const* d_in, const int* in_sizes, int n_in,
                              void* d_out, int out_size, void* d_ws, size_t ws_size,
                              hipStream_t stream) {
  const float* input_seq = (const float*)d_in[0];
  // d_in[1] = future_n (== 128, fixed problem size)
  const float* W_enc  = (const float*)d_in[2];
  const float* b_enc  = (const float*)d_in[3];
  const float* W_ih   = (const float*)d_in[4];
  const float* b_ih   = (const float*)d_in[5];
  const float* W_hh   = (const float*)d_in[6];
  const float* b_hh   = (const float*)d_in[7];
  const float* W_fenc = (const float*)d_in[8];
  const float* b_fenc = (const float*)d_in[9];
  const float* W_dec  = (const float*)d_in[10];
  const float* b_dec  = (const float*)d_in[11];
  float* out = (float*)d_out;

  // workspace carve-up (256B aligned)
  char* ws = (char*)d_ws;
  size_t off = 0;
  auto take = [&](size_t bytes) {
    void* p = ws + off;
    off = (off + bytes + 255) & ~(size_t)255;
    return p;
  };
  bf16_t* Wih_bf   = (bf16_t*)take((size_t)G_ * RIP_ * 2);
  bf16_t* Whh_bf   = (bf16_t*)take((size_t)G_ * HID_ * 2);
  bf16_t* Wenc_bf  = (bf16_t*)take((size_t)RIP_ * FIN_ * 2);
  bf16_t* Wfenc_bf = (bf16_t*)take((size_t)RIP_ * HID_ * 2);
  bf16_t* Wdec_bf  = (bf16_t*)take((size_t)FOUT_ * HID_ * 2);
  float*  bsum     = (float*)take((size_t)G_ * 4);
  float*  cbuf     = (float*)take((size_t)B_ * HID_ * 4);
  bf16_t* h0       = (bf16_t*)take((size_t)B_ * HID_ * 2);
  bf16_t* ubf      = (bf16_t*)take((size_t)B_ * RIP_ * 2);
  bf16_t* enc_bf   = (bf16_t*)take((size_t)B_ * T_ * RIP_ * 2);
  bf16_t* hs       = (bf16_t*)take((size_t)(T_ + FUT_) * B_ * HID_ * 2);
  (void)ws_size; (void)in_sizes; (void)n_in; (void)out_size;

  dim3 blk(256);

  prep_kernel<<<2048, blk, 0, stream>>>(W_enc, W_ih, W_hh, W_fenc, W_dec,
                                        b_ih, b_hh, Wenc_bf, Wih_bf, Whh_bf,
                                        Wfenc_bf, Wdec_bf, bsum, cbuf, h0);

  enc_kernel<<<8192, blk, 0, stream>>>(input_seq, Wenc_bf, b_enc, enc_bf);

  const bf16_t* h_prev = h0;
  for (int t = 0; t < T_; ++t) {
    bf16_t* hs_t = hs + (size_t)t * B_ * HID_;
    lstm_step_kernel<<<32, blk, 0, stream>>>(enc_bf + (size_t)t * RIP_,
                                             T_ * RIP_, Wih_bf, Whh_bf, bsum,
                                             cbuf, h_prev, hs_t);
    h_prev = hs_t;
  }
  for (int ft = 0; ft < FUT_; ++ft) {
    int t = T_ + ft;
    fenc_kernel<<<16, blk, 0, stream>>>(h_prev, Wfenc_bf, b_fenc, ubf);
    bf16_t* hs_t = hs + (size_t)t * B_ * HID_;
    lstm_step_kernel<<<32, blk, 0, stream>>>(ubf, RIP_, Wih_bf, Whh_bf, bsum,
                                             cbuf, h_prev, hs_t);
    h_prev = hs_t;
  }

  dec_kernel<<<2560, blk, 0, stream>>>(hs, Wdec_bf, b_dec, out);
}
